// MSAColumnAttention_16655883174275
// MI455X (gfx1250) — matrix operand order repaired and checked by reading, not compiled, not run on hardware
//
#include <hip/hip_runtime.h>
#include <hip/hip_bf16.h>
#include <math.h>

typedef unsigned int u32;
typedef __bf16 bf16_t;
typedef __attribute__((ext_vector_type(16))) __bf16 v16bf;
typedef __attribute__((ext_vector_type(8)))  float  v8f;
typedef __attribute__((ext_vector_type(4)))  u32    u32x4;

#define S_SEQ 512
#define R_RES 256
#define CM    256
#define CH    32
#define NH    8
#define EPSV  1e-5f
#define SCALE 0.17677669529663687f  // 1/sqrt(32)

// padded LDS strides (avoid 16-lane bank collisions on fragment loads)
#define QSTR 40    // Q/K/G rows: [512][40] bf16
#define VSTR 520   // V^T rows:   [32][520] bf16
#define XSTR 264   // X staging / A-tile rows (k padded 256->264)
#define WSTR 264   // W^T rows (k padded 256->264)
#define PSTR 40    // P staging rows

union FragU  { v16bf v; u32x4 q[2]; };
union Pack8  { bf16_t h[8]; u32x4 q; };
union PackF8 { float f[8]; u32x4 q[2]; };

__device__ __forceinline__ v8f zero8() {
  v8f z = {0.f,0.f,0.f,0.f,0.f,0.f,0.f,0.f};
  return z;
}

// A fragment (16x32 bf16, M x K). Element (m,K) at base[m*rowStride + kOff + K].
// lanes 0-15: M=lane, K in {0..7,16..23}; lanes 16-31: K in {8..15,24..31}.
__device__ __forceinline__ v16bf load_frag_a(const bf16_t* base, int rowStride, int kOff, int lane) {
  int m  = lane & 15;
  int kb = (lane & 16) ? 8 : 0;
  const bf16_t* p = base + m * rowStride + kOff + kb;
  FragU f;
  f.q[0] = *(const u32x4*)(p);        // K pairs +0..7
  f.q[1] = *(const u32x4*)(p + 16);   // K pairs +16..23
  return f.v;
}

// B fragment (32x16 bf16, K x N). Element (k,n) at base[n*nStride + kOff + k].
// lanes 0-15: N=lane, K in {0..7,16..23}; lanes 16-31: K in {8..15,24..31}.
__device__ __forceinline__ v16bf load_frag_b(const bf16_t* base, int nStride, int kOff, int lane) {
  int n  = lane & 15;
  int kb = (lane & 16) ? 8 : 0;
  const bf16_t* p = base + n * nStride + kOff + kb;
  FragU f;
  f.q[0] = *(const u32x4*)(p);
  f.q[1] = *(const u32x4*)(p + 16);
  return f.v;
}

__device__ __forceinline__ v8f wmma_bf16(v16bf a, v16bf b, v8f c) {
  return __builtin_amdgcn_wmma_f32_16x16x32_bf16(false, a, false, b, (short)0, c, false, false);
}

__device__ __forceinline__ float wave_sum(float v) {
#pragma unroll
  for (int m = 16; m >= 1; m >>= 1) v += __shfl_xor(v, m, 32);
  return v;
}

// One workgroup per (residue r, head h). 256 threads = 8 waves.
__global__ __launch_bounds__(256, 1)
void msa_attn_head_kernel(const float* __restrict__ msa,
                          const float* __restrict__ ln_w, const float* __restrict__ ln_b,
                          const float* __restrict__ wq, const float* __restrict__ wk,
                          const float* __restrict__ wv, const float* __restrict__ wg,
                          bf16_t* __restrict__ o_heads) {
  extern __shared__ char smem[];
  bf16_t* Qs  = (bf16_t*)(smem);                     // [512][QSTR]  40960 B
  bf16_t* Ks  = (bf16_t*)(smem + 40960);             // [512][QSTR]
  bf16_t* Gs  = (bf16_t*)(smem + 81920);             // [512][QSTR]
  bf16_t* Vts = (bf16_t*)(smem + 122880);            // [32][VSTR]   33280 B
  char* pa = smem + 156160;                          // phase-A region (reused in B)
  bf16_t* Wt  = (bf16_t*)(pa);                       // [4][32][WSTR] 67584 B
  float*  LNW = (float*)(pa + 67584);                // [8][256]
  float*  LNB = (float*)(pa + 75776);                // [8][256]
  bf16_t* Xst = (bf16_t*)(pa + 83968);               // [8 waves][16][XSTR] 67584 B
  float*  Mst = (float*)(pa);                        // [512]        (phase B)
  float*  Dinv= (float*)(pa + 2048);                 // [512]        (phase B)
  bf16_t* Pst = (bf16_t*)(pa + 4096);                // [8 waves][16][PSTR] (phase B)

  const int tid  = threadIdx.x;
  const int lane = tid & 31;
  const int wave = tid >> 5;
  const int r = blockIdx.x & (R_RES - 1);
  const int h = blockIdx.x >> 8;

  // ---- stage transposed bf16 weights + LN params ----
  const float* Wsrc[4] = {wq, wk, wv, wg};
#pragma unroll
  for (int mat = 0; mat < 4; ++mat) {
    const float* w = Wsrc[mat] + h * (CM * CH);
    for (int idx = tid; idx < CM * CH; idx += 256) {
      int k = idx >> 5, c = idx & 31;
      Wt[mat * (32 * WSTR) + c * WSTR + k] = (bf16_t)w[idx];   // Wt[mat][c'][k]
    }
  }
  for (int idx = tid; idx < NH * CM; idx += 256) {
    LNW[idx] = ln_w[idx];
    LNB[idx] = ln_b[idx];
  }
  __syncthreads();

  // ---- phase A: LN chain (heads 0..h) + Q/K/V/G projections via WMMA ----
  {
    const int n  = lane & 15;                 // row_local in D layout
    const int m0 = (lane & 16) ? 8 : 0;       // channel sub-block in D layout
    bf16_t* xw = Xst + wave * (16 * XSTR);
#pragma clang loop unroll(disable)
    for (int t = wave * 4; t < wave * 4 + 4; ++t) {
#pragma clang loop unroll(disable)
      for (int rr = 0; rr < 16; ++rr) {
        int s = t * 16 + rr;
        const float* xr = msa + ((size_t)s * R_RES + r) * CM + lane * 8;
        float xv[8];
        const float4 x0 = *(const float4*)xr;
        const float4 x1 = *(const float4*)(xr + 4);
        xv[0]=x0.x; xv[1]=x0.y; xv[2]=x0.z; xv[3]=x0.w;
        xv[4]=x1.x; xv[5]=x1.y; xv[6]=x1.z; xv[7]=x1.w;
#pragma clang loop unroll(disable)
        for (int i = 0; i <= h; ++i) {
          float sm = 0.f;
#pragma unroll
          for (int j = 0; j < 8; ++j) sm += xv[j];
          float mu = wave_sum(sm) * (1.0f / CM);
          float vs = 0.f;
#pragma unroll
          for (int j = 0; j < 8; ++j) { float d = xv[j] - mu; vs += d * d; }
          float var = wave_sum(vs) * (1.0f / CM);
          float rstd = rsqrtf(var + EPSV);
          const float* wrow = LNW + i * CM + lane * 8;
          const float* brow = LNB + i * CM + lane * 8;
#pragma unroll
          for (int j = 0; j < 8; ++j) xv[j] = (xv[j] - mu) * rstd * wrow[j] + brow[j];
        }
        Pack8 px;
#pragma unroll
        for (int j = 0; j < 8; ++j) px.h[j] = (bf16_t)xv[j];
        *(u32x4*)(xw + rr * XSTR + lane * 8) = px.q;
      }
      const int row = t * 16 + n;
      // D = W^T(A) x X^T(B): lane = row, VGPR elem = output channel (contiguous)
#pragma clang loop unroll(disable)
      for (int mat = 0; mat < 4; ++mat) {
        v8f a0 = zero8(), a1 = zero8();   // channels 0..15 / 16..31
        const bf16_t* wb = Wt + mat * (32 * WSTR);
#pragma clang loop unroll(disable)
        for (int kb = 0; kb < 8; ++kb) {
          v16bf bx = load_frag_b(xw, XSTR, kb * 32, lane);
          v16bf w0 = load_frag_a(wb,             WSTR, kb * 32, lane);
          v16bf w1 = load_frag_a(wb + 16 * WSTR, WSTR, kb * 32, lane);
          a0 = wmma_bf16(w0, bx, a0);
          a1 = wmma_bf16(w1, bx, a1);
        }
        if (mat == 2) {
#pragma unroll
          for (int e = 0; e < 8; ++e) {
            Vts[(m0 + e) * VSTR + row]      = (bf16_t)a0[e];
            Vts[(16 + m0 + e) * VSTR + row] = (bf16_t)a1[e];
          }
        } else {
          bf16_t* dst = (mat == 0) ? Qs : ((mat == 1) ? Ks : Gs);
          Pack8 p0, p1;
          if (mat == 3) {
#pragma unroll
            for (int e = 0; e < 8; ++e) {
              p0.h[e] = (bf16_t)(1.0f / (1.0f + __expf(-a0[e])));
              p1.h[e] = (bf16_t)(1.0f / (1.0f + __expf(-a1[e])));
            }
          } else {
#pragma unroll
            for (int e = 0; e < 8; ++e) {
              p0.h[e] = (bf16_t)a0[e];
              p1.h[e] = (bf16_t)a1[e];
            }
          }
          *(u32x4*)(dst + row * QSTR + m0)      = p0.q;
          *(u32x4*)(dst + row * QSTR + 16 + m0) = p1.q;
        }
      }
    }
  }
  __syncthreads();

  // ---- pass 1: per-column (query-axis) softmax stats ----
  {
    const int n = lane & 15;
#pragma clang loop unroll(disable)
    for (int jt = wave * 4; jt < wave * 4 + 4; ++jt) {
      float mx = -3.0e38f, dn = 0.f;
#pragma clang loop unroll(disable)
      for (int it = 0; it < 32; ++it) {
        v16bf a = load_frag_a(Qs + it * 16 * QSTR, QSTR, 0, lane);
        v16bf b = load_frag_b(Ks + jt * 16 * QSTR, QSTR, 0, lane);
        v8f sA = wmma_bf16(a, b, zero8());
        float vv[8];
#pragma unroll
        for (int e = 0; e < 8; ++e) vv[e] = sA[e] * SCALE;
        float tm = fmaxf(fmaxf(fmaxf(vv[0], vv[1]), fmaxf(vv[2], vv[3])),
                         fmaxf(fmaxf(vv[4], vv[5]), fmaxf(vv[6], vv[7])));
        float nm = fmaxf(mx, tm);
        float add = 0.f;
#pragma unroll
        for (int e = 0; e < 8; ++e) add += __expf(vv[e] - nm);
        dn = dn * __expf(mx - nm) + add;
        mx = nm;
      }
      float om = __shfl_xor(mx, 16, 32);
      float od = __shfl_xor(dn, 16, 32);
      float nm = fmaxf(mx, om);
      float d  = dn * __expf(mx - nm) + od * __expf(om - nm);
      if (lane < 16) {
        Mst[jt * 16 + n]  = nm;
        Dinv[jt * 16 + n] = 1.0f / d;
      }
    }
  }
  __syncthreads();

  // ---- fold 1/d_j into V ----
  for (int idx = tid; idx < 32 * 512; idx += 256) {
    int c = idx >> 9, j = idx & 511;
    Vts[c * VSTR + j] = (bf16_t)((float)Vts[c * VSTR + j] * Dinv[j]);
  }
  __syncthreads();

  // ---- pass 2: O^T = V'^T(A) @ exp(S-m)^T(B), gate, store ----
  {
    const int n  = lane & 15;
    const int m0 = (lane & 16) ? 8 : 0;
    bf16_t* pw = Pst + wave * (16 * PSTR);
#pragma clang loop unroll(disable)
    for (int it = wave * 4; it < wave * 4 + 4; ++it) {
      v8f o0 = zero8(), o1 = zero8();     // channels 0..15 / 16..31
      v16bf aq = load_frag_a(Qs + it * 16 * QSTR, QSTR, 0, lane);
#pragma clang loop unroll(disable)
      for (int jp = 0; jp < 16; ++jp) {
#pragma clang loop unroll(disable)
        for (int half = 0; half < 2; ++half) {
          int jt = jp * 2 + half;
          v16bf bk = load_frag_b(Ks + jt * 16 * QSTR, QSTR, 0, lane);
          v8f sA = wmma_bf16(aq, bk, zero8());   // lane=j_local, elems=i rows
          float mj = Mst[jt * 16 + n];
#pragma unroll
          for (int e = 0; e < 8; ++e)
            pw[(m0 + e) * PSTR + half * 16 + n] = (bf16_t)__expf(sA[e] * SCALE - mj);
        }
        v16bf av0 = load_frag_a(Vts,             VSTR, jp * 32, lane);
        v16bf av1 = load_frag_a(Vts + 16 * VSTR, VSTR, jp * 32, lane);
        v16bf bp  = load_frag_b(pw, PSTR, 0, lane);
        o0 = wmma_bf16(av0, bp, o0);
        o1 = wmma_bf16(av1, bp, o1);
      }
      const int row = it * 16 + n;
      Pack8 g0 = *(const Pack8*)(Gs + row * QSTR + m0);
      Pack8 g1 = *(const Pack8*)(Gs + row * QSTR + 16 + m0);
      Pack8 r0, r1;
#pragma unroll
      for (int e = 0; e < 8; ++e) {
        r0.h[e] = (bf16_t)(o0[e] * (float)g0.h[e]);
        r1.h[e] = (bf16_t)(o1[e] * (float)g1.h[e]);
      }
      bf16_t* op = o_heads + ((size_t)(r * 512 + row)) * 256 + h * 32;
      *(u32x4*)(op + m0)      = r0.q;
      *(u32x4*)(op + 16 + m0) = r1.q;
    }
  }
}

// Output projection: [131072 x 256] bf16 @ out_w[256 x 256] + bias -> out[s][r][256] f32
__global__ __launch_bounds__(256, 1)
void out_proj_kernel(const bf16_t* __restrict__ o_heads,
                     const float* __restrict__ out_w,
                     const float* __restrict__ out_b,
                     float* __restrict__ out) {
  extern __shared__ char smem[];
  bf16_t* Wt2 = (bf16_t*)(smem);             // [256 n][WSTR k]  135168 B
  bf16_t* At  = (bf16_t*)(smem + 135168);    // [16][XSTR]       8448 B
  float*  Bs  = (float*)(smem + 143616);     // [256]
  const int tid = threadIdx.x, lane = tid & 31, wave = tid >> 5;

  for (int idx = tid; idx < 256 * 256; idx += 256) {
    int k = idx >> 8, n2 = idx & 255;
    Wt2[n2 * WSTR + k] = (bf16_t)out_w[idx];
  }
  if (tid < 256) Bs[tid] = out_b[tid];
  __syncthreads();

  const int n  = lane & 15;
  const int m0 = (lane & 16) ? 8 : 0;
  for (int mt = blockIdx.x; mt < (R_RES * S_SEQ) / 16; mt += gridDim.x) {
    const u32* src = (const u32*)(o_heads + (size_t)mt * 16 * 256);
    for (int idx = tid; idx < 2048; idx += 256) {
      int row = idx >> 7, kd = idx & 127;
      *((u32*)(At + row * XSTR) + kd) = src[idx];
    }
    __syncthreads();
    const int row0 = mt * 16;
    const int r  = row0 >> 9;     // flat row = r*512 + s
    const int s0 = row0 & 511;
    const int s  = s0 + n;
#pragma clang loop unroll(disable)
    for (int ct = wave * 2; ct < wave * 2 + 2; ++ct) {
      v8f acc = zero8();
#pragma clang loop unroll(disable)
      for (int kb = 0; kb < 8; ++kb) {
        v16bf a = load_frag_a(Wt2 + ct * 16 * WSTR, WSTR, kb * 32, lane);
        v16bf b = load_frag_b(At, XSTR, kb * 32, lane);
        acc = wmma_bf16(a, b, acc);   // lane = row(s), elems = cols (contiguous)
      }
      const int col0 = ct * 16 + m0;
      PackF8 o;
#pragma unroll
      for (int e = 0; e < 8; ++e) o.f[e] = acc[e] + Bs[col0 + e];
      float* dst = out + ((size_t)s * R_RES + r) * CM + col0;
      *(u32x4*)(dst)     = o.q[0];
      *(u32x4*)(dst + 4) = o.q[1];
    }
    __syncthreads();
  }
}

extern "C" void kernel_launch(void* const* d_in, const int* in_sizes, int n_in,
                              void* d_out, int out_size, void* d_ws, size_t ws_size,
                              hipStream_t stream) {
  const float* msa  = (const float*)d_in[0];
  const float* ln_w = (const float*)d_in[1];
  const float* ln_b = (const float*)d_in[2];
  const float* wq   = (const float*)d_in[3];
  const float* wk   = (const float*)d_in[4];
  const float* wv   = (const float*)d_in[5];
  const float* wg   = (const float*)d_in[6];
  const float* ow   = (const float*)d_in[7];
  const float* ob   = (const float*)d_in[8];
  float* out = (float*)d_out;
  bf16_t* o_heads = (bf16_t*)d_ws;  // [R][S][H*C] bf16 = 64 MB

  const size_t sh1 = 307712;  // attention kernel dynamic LDS
  const size_t sh2 = 144640;  // projection kernel dynamic LDS
  (void)hipFuncSetAttribute((const void*)msa_attn_head_kernel,
                            hipFuncAttributeMaxDynamicSharedMemorySize, (int)sh1);
  (void)hipFuncSetAttribute((const void*)out_proj_kernel,
                            hipFuncAttributeMaxDynamicSharedMemorySize, (int)sh2);

  msa_attn_head_kernel<<<dim3(R_RES * NH), dim3(256), sh1, stream>>>(
      msa, ln_w, ln_b, wq, wk, wv, wg, o_heads);
  out_proj_kernel<<<dim3(256), dim3(256), sh2, stream>>>(o_heads, ow, ob, out);
}
